// CreateConvlstmModel_4449586119146
// MI455X (gfx1250) — matrix operand order repaired and checked by reading, not compiled, not run on hardware
//
#include <hip/hip_runtime.h>
#include <math.h>

typedef __attribute__((ext_vector_type(16))) _Float16 v16h;
typedef __attribute__((ext_vector_type(8)))  _Float16 v8h;
typedef __attribute__((ext_vector_type(8)))  float    v8f;

__device__ __forceinline__ float hsig(float x) {
    return fminf(fmaxf(0.2f * x + 0.5f, 0.0f), 1.0f);
}

// ---------------------------------------------------------------------------
// One-time per layer: build the fused implicit-GEMM B matrix in f16,
// transposed to [n][k] and zero-padded to KP = ceil((9*CIN+9*F)/32)*32.
// k < 9*CIN -> Wx patch; else -> Wh patch.
// ---------------------------------------------------------------------------
template<int CIN, int F>
__global__ void prep_weights(const float* __restrict__ Wx,
                             const float* __restrict__ Wh,
                             _Float16* __restrict__ out)
{
    constexpr int NG = 4 * F;
    constexpr int KX = 9 * CIN;
    constexpr int KT = KX + 9 * F;
    constexpr int KP = (KT + 31) & ~31;
    const int idx = blockIdx.x * blockDim.x + threadIdx.x;
    if (idx >= NG * KP) return;
    const int n = idx / KP, k = idx - (idx / KP) * KP;
    float v = 0.0f;
    if (k < KX)      v = Wx[k * NG + n];
    else if (k < KT) v = Wh[(k - KX) * NG + n];
    out[n * KP + k] = (_Float16)v;
}

// ---------------------------------------------------------------------------
// Fused ConvLSTM step:  g = conv_valid(x_t, Wx) + conv_same(h_prev, Wh) + b
// Grid: (ceil(Wo/32), Ho, B) -> no runtime division anywhere in the kernel.
// Block = 64 threads = 2 wave32 waves; each wave owns a 16-row M tile (rows
// are 16 consecutive wo positions) across all 4F/16 N tiles.
// ---------------------------------------------------------------------------
template<int CIN, int F>
__global__ __launch_bounds__(64)
void convlstm_step(const float* __restrict__ x,      // (B,T,H,W,CIN)
                   const _Float16* __restrict__ Bw,  // [4F][KP] fused f16 weights
                   const float* __restrict__ bias,   // (4F)
                   const float* __restrict__ h_prev, // (B,Ho,Wo,F)
                   float* __restrict__ h_next,       // (B,Ho,Wo,F)
                   float* __restrict__ c,            // (B,Ho,Wo,F) in/out
                   float* __restrict__ hs,           // (B,T,Ho,Wo,F)
                   int t, int T, int H, int W, int Ho, int Wo)
{
    constexpr int NG = 4 * F;                 // gate channels
    constexpr int KX = 9 * CIN;               // input-conv K
    constexpr int KT = KX + 9 * F;            // fused K
    constexpr int KP = (KT + 31) & ~31;       // padded to 32
    constexpr int NT = NG / 16;               // N tiles

    __shared__ alignas(32) _Float16 Alds[32 * KP];   // 32 wo-rows x KP
    __shared__ alignas(32) _Float16 Blds[NG * KP];   // [n][k]
    __shared__ alignas(32) float    Glds[32 * NG];   // staged gates

    const int tid = threadIdx.x;
    const int wo0 = blockIdx.x * 32;
    const int ho  = blockIdx.y;
    const int b   = blockIdx.z;

    // ---- stage pre-transposed f16 weights: pure 16B vector copy ----
    {
        const v8h* __restrict__ src = (const v8h*)Bw;
        v8h* __restrict__ dst = (v8h*)Blds;
        for (int idx = tid; idx < (NG * KP) / 8; idx += 64) dst[idx] = src[idx];
    }

    // ---- gather A tile: 32 wo-rows of the fused im2col patch ----
    for (int idx = tid; idx < 32 * KP; idx += 64) {
        const int row = idx / KP, k = idx - (idx / KP) * KP;   // k fastest:
        const int wo = wo0 + row;                              // conflict-free LDS
        float v = 0.0f;
        if (wo < Wo && k < KT) {
            if (k < KX) {
                const int ci = k % CIN, p = k / CIN;
                const int kw = p % 3,   kh = p / 3;
                v = x[(((size_t)(b * T + t) * H + (ho + kh)) * W + (wo + kw)) * CIN + ci];
                // Pull next timestep's patch toward the WGP (speculative,
                // safe per ISA 10.5). Emits global_prefetch_b8.
                if (k == 0 && t + 1 < T)
                    __builtin_prefetch(
                        &x[(((size_t)(b * T + t + 1) * H + ho) * W + wo) * CIN], 0, 1);
            } else {
                const int kk = k - KX;
                const int f  = kk % F, p = kk / F;
                const int kw = p % 3,  kh = p / 3;
                const int y  = ho + kh - 1, xw = wo + kw - 1;
                if (y >= 0 && y < Ho && xw >= 0 && xw < Wo)
                    v = h_prev[((size_t)(b * Ho + y) * Wo + xw) * F + kk % F];
            }
        }
        Alds[row * KP + k] = (_Float16)v;
    }
    __syncthreads();

    // ---- WMMA: each wave owns a 16-row M tile, all NT column tiles ----
    const int wave = tid >> 5;
    const int lane = tid & 31;
    const int l15  = lane & 15;
    const bool hi  = (lane >= 16);
    const int rowb = wave * 16;

    v8f acc[NT] = {};

#pragma unroll
    for (int kc = 0; kc < KP; kc += 32) {
        // A fragment (16-bit 16x32): lanes 0-15 -> K kc..kc+15,
        // lanes 16-31 -> K kc+16..kc+31; row M = lane&15.
        const _Float16* arow = &Alds[(rowb + l15) * KP + kc + (hi ? 16 : 0)];
        v8h alo = *(const v8h*)(arow);
        v8h ahi = *(const v8h*)(arow + 8);
        v16h a = __builtin_shufflevector(alo, ahi,
                                         0, 1, 2, 3, 4, 5, 6, 7,
                                         8, 9, 10, 11, 12, 13, 14, 15);
#pragma unroll
        for (int n = 0; n < NT; ++n) {
            // B fragment (32x16): lane holds column N=lane&15, 16 contiguous
            // K values at kc (lanes<16) / kc+16 (lanes>=16).
            const _Float16* bcol = &Blds[(n * 16 + l15) * KP + kc + (hi ? 16 : 0)];
            v16h bf = *(const v16h*)(bcol);
            acc[n] = __builtin_amdgcn_wmma_f32_16x16x32_f16(
                false, a, false, bf, (short)0, acc[n], false, false);
        }
    }

    // ---- stage gate tiles to LDS (C/D f32 16x16 layout) ----
#pragma unroll
    for (int n = 0; n < NT; ++n) {
#pragma unroll
        for (int r = 0; r < 8; ++r) {
            const int m = (hi ? 8 : 0) + r;
            Glds[(rowb + m) * NG + n * 16 + l15] = acc[n][r];
        }
    }
    __syncthreads();

    // ---- gate math: i,f,c,o split along channels; update c, h ----
    const int HoWo = Ho * Wo;
    for (int idx = tid; idx < 32 * F; idx += 64) {
        const int row = idx / F, f = idx - (idx / F) * F;
        const int wo = wo0 + row;
        if (wo >= Wo) continue;
        const float gi = Glds[row * NG + 0 * F + f] + bias[0 * F + f];
        const float gf = Glds[row * NG + 1 * F + f] + bias[1 * F + f];
        const float gc = Glds[row * NG + 2 * F + f] + bias[2 * F + f];
        const float go = Glds[row * NG + 3 * F + f] + bias[3 * F + f];
        const size_t ci = ((size_t)(b * Ho + ho) * Wo + wo) * F + f;
        const float cn = hsig(gf) * c[ci] + hsig(gi) * tanhf(gc);
        c[ci] = cn;
        const float hv = hsig(go) * tanhf(cn);
        h_next[ci] = hv;
        hs[((size_t)(b * T + t) * HoWo + ho * Wo + wo) * F + f] = hv;
    }
}

// ---------------------------------------------------------------------------
// MaxPooling over H,W with pool 2x2 stride 2 padding SAME on (BT,H,W,C).
// ---------------------------------------------------------------------------
__global__ void maxpool2x2(const float* __restrict__ in, float* __restrict__ out,
                           int BT, int H, int W, int C, int Hp, int Wp)
{
    const int idx = blockIdx.x * blockDim.x + threadIdx.x;
    const int total = BT * Hp * Wp * C;
    if (idx >= total) return;
    int r = idx;
    const int ch = r % C;  r /= C;
    const int wp = r % Wp; r /= Wp;
    const int hp = r % Hp; r /= Hp;   // r = bt
    const int h0 = hp * 2, w0 = wp * 2;
    float m = -INFINITY;
#pragma unroll
    for (int dy = 0; dy < 2; ++dy)
#pragma unroll
        for (int dx = 0; dx < 2; ++dx) {
            const int y = h0 + dy, xw = w0 + dx;
            if (y < H && xw < W)
                m = fmaxf(m, in[(((size_t)r * H + y) * W + xw) * C + ch]);
        }
    out[idx] = m;
}

__global__ void zero_f32(float* __restrict__ p, int n)
{
    const int i = blockIdx.x * blockDim.x + threadIdx.x;
    if (i < n) p[i] = 0.0f;
}

// ---------------------------------------------------------------------------
// Dense (32x50, K=18816) + softmax. Tiny: plain VALU. One block per batch row.
// ---------------------------------------------------------------------------
__global__ __launch_bounds__(64)
void dense_softmax(const float* __restrict__ flat, const float* __restrict__ Wd,
                   const float* __restrict__ bd, float* __restrict__ out,
                   int K, int N)
{
    __shared__ float logit[64];
    __shared__ float red[2];
    const int b = blockIdx.x, tx = threadIdx.x;
    float acc = 0.0f;
    if (tx < N) {
        const float* xr = flat + (size_t)b * K;
#pragma unroll 4
        for (int k = 0; k < K; ++k) acc += xr[k] * Wd[(size_t)k * N + tx];
        acc += bd[tx];
    }
    logit[tx] = (tx < N) ? acc : -INFINITY;
    __syncthreads();
    if (tx == 0) {
        float mx = -INFINITY;
        for (int i = 0; i < N; ++i) mx = fmaxf(mx, logit[i]);
        red[0] = mx;
    }
    __syncthreads();
    const float e = (tx < N) ? expf(acc - red[0]) : 0.0f;
    logit[tx] = e;
    __syncthreads();
    if (tx == 0) {
        float s = 0.0f;
        for (int i = 0; i < N; ++i) s += logit[i];
        red[1] = s;
    }
    __syncthreads();
    if (tx < N) out[(size_t)b * N + tx] = e / red[1];
}

// ---------------------------------------------------------------------------
// Host orchestration. Inputs (setup_inputs order):
//  0:x 1:Wx1 2:Wh1 3:b1 4:Wx2 5:Wh2 6:b2 7:Wx3 8:Wh3 9:b3 10:Wx4 11:Wh4 12:b4
//  13:Wd 14:bd
// Workspace (floats): hs(48,771,072) | pool(12,192,768) | hA|hB|c (2,032,128 ea)
//  then 61,440 bytes of fused f16 weights. Total ~268 MB.
// ---------------------------------------------------------------------------
extern "C" void kernel_launch(void* const* d_in, const int* in_sizes, int n_in,
                              void* d_out, int out_size, void* d_ws, size_t ws_size,
                              hipStream_t stream)
{
    (void)in_sizes; (void)n_in; (void)out_size; (void)ws_size;

    const float* x     = (const float*)d_in[0];
    const float* Wx[4] = {(const float*)d_in[1], (const float*)d_in[4],
                          (const float*)d_in[7], (const float*)d_in[10]};
    const float* Wh[4] = {(const float*)d_in[2], (const float*)d_in[5],
                          (const float*)d_in[8], (const float*)d_in[11]};
    const float* bs[4] = {(const float*)d_in[3], (const float*)d_in[6],
                          (const float*)d_in[9], (const float*)d_in[12]};
    const float* Wd = (const float*)d_in[13];
    const float* bd = (const float*)d_in[14];

    const int B = 32, T = 24;
    const int Hs[4] = {128, 63, 31, 15};
    const int Fs[4] = {4, 8, 12, 16};
    // fused B-matrix sizes in f16 elements: NG*KP
    const int WBsz[4] = {16 * 64, 32 * 128, 48 * 192, 64 * 256};

    float* ws = (float*)d_ws;
    const size_t HS_CAP = (size_t)B * T * 126 * 126 * 4;  // 48,771,072
    const size_t PO_CAP = (size_t)B * T * 63 * 63 * 4;    // 12,192,768
    const size_t HC_CAP = (size_t)B * 126 * 126 * 4;      //  2,032,128
    float* hsbuf = ws;
    float* pool  = hsbuf + HS_CAP;
    float* hA    = pool + PO_CAP;
    float* hB    = hA + HC_CAP;
    float* cB    = hB + HC_CAP;
    _Float16* wb[4];
    {
        _Float16* wbase = (_Float16*)(cB + HC_CAP);
        wb[0] = wbase;
        wb[1] = wb[0] + WBsz[0];
        wb[2] = wb[1] + WBsz[1];
        wb[3] = wb[2] + WBsz[2];
    }

    // Build fused f16 weight matrices once.
    prep_weights<3, 4>  <<<(WBsz[0] + 255) / 256, 256, 0, stream>>>(Wx[0], Wh[0], wb[0]);
    prep_weights<4, 8>  <<<(WBsz[1] + 255) / 256, 256, 0, stream>>>(Wx[1], Wh[1], wb[1]);
    prep_weights<8, 12> <<<(WBsz[2] + 255) / 256, 256, 0, stream>>>(Wx[2], Wh[2], wb[2]);
    prep_weights<12, 16><<<(WBsz[3] + 255) / 256, 256, 0, stream>>>(Wx[3], Wh[3], wb[3]);

    const float* cur_in = x;
    for (int L = 0; L < 4; ++L) {
        const int H  = Hs[L], W = Hs[L];
        const int F  = Fs[L];
        const int Ho = H - 2, Wo = W - 2;
        const int hcn = B * Ho * Wo * F;

        zero_f32<<<(hcn + 255) / 256, 256, 0, stream>>>(hA, hcn);
        zero_f32<<<(hcn + 255) / 256, 256, 0, stream>>>(cB, hcn);

        float* hp = hA;
        float* hn = hB;
        const dim3 grid((Wo + 31) / 32, Ho, B);
        for (int t = 0; t < T; ++t) {
            switch (L) {
            case 0: convlstm_step<3, 4><<<grid, 64, 0, stream>>>(
                        cur_in, wb[0], bs[0], hp, hn, cB, hsbuf, t, T, H, W, Ho, Wo); break;
            case 1: convlstm_step<4, 8><<<grid, 64, 0, stream>>>(
                        cur_in, wb[1], bs[1], hp, hn, cB, hsbuf, t, T, H, W, Ho, Wo); break;
            case 2: convlstm_step<8, 12><<<grid, 64, 0, stream>>>(
                        cur_in, wb[2], bs[2], hp, hn, cB, hsbuf, t, T, H, W, Ho, Wo); break;
            default: convlstm_step<12, 16><<<grid, 64, 0, stream>>>(
                        cur_in, wb[3], bs[3], hp, hn, cB, hsbuf, t, T, H, W, Ho, Wo); break;
            }
            float* tmp = hp; hp = hn; hn = tmp;
        }

        const int Hp = (Ho + 1) / 2, Wp = (Wo + 1) / 2;
        const int tot = B * T * Hp * Wp * F;
        maxpool2x2<<<(tot + 255) / 256, 256, 0, stream>>>(
            hsbuf, pool, B * T, Ho, Wo, F, Hp, Wp);
        cur_in = pool;
    }

    dense_softmax<<<32, 64, 0, stream>>>(pool, Wd, bd, (float*)d_out, 18816, 50);
}
